// FRI_GINConvNet_8108898255299
// MI455X (gfx1250) — compile-verified, matrix operand order checked by reading, not run on hardware
//
#include <hip/hip_runtime.h>
#include <hip/hip_bf16.h>

typedef __attribute__((ext_vector_type(16))) __bf16 v16bf;
typedef __attribute__((ext_vector_type(8)))  __bf16 v8bf;
typedef __attribute__((ext_vector_type(8)))  float  v8f;

#define BN_EPS 1e-5f

// ---------------------------------------------------------------- utilities
__global__ void k_zero(float* __restrict__ p, long long n) {
  long long stride = (long long)gridDim.x * blockDim.x;
  for (long long i = (long long)blockIdx.x * blockDim.x + threadIdx.x; i < n; i += stride)
    p[i] = 0.0f;
}

// agg[dst[e]][f] += h[src[e]][f]  (float atomics resolved in L2)
__global__ void k_scatter_add(const float* __restrict__ h,
                              const long long* __restrict__ src,
                              const long long* __restrict__ dst,
                              float* __restrict__ agg, long long E, int F) {
  long long total  = E * (long long)F;
  long long stride = (long long)gridDim.x * blockDim.x;
  for (long long idx = (long long)blockIdx.x * blockDim.x + threadIdx.x; idx < total; idx += stride) {
    long long e = idx / F;
    int f = (int)(idx - e * (long long)F);
    long long s = src[e], d = dst[e];
    atomicAdd(&agg[d * (long long)F + f], h[s * (long long)F + f]);
  }
}

// g[batch[i]][f] += h[i][f]
__global__ void k_pool(const float* __restrict__ h, const long long* __restrict__ batch,
                       float* __restrict__ g, long long Nn, int F) {
  long long total  = Nn * (long long)F;
  long long stride = (long long)gridDim.x * blockDim.x;
  for (long long idx = (long long)blockIdx.x * blockDim.x + threadIdx.x; idx < total; idx += stride) {
    long long i = idx / F;
    int f = (int)(idx - i * (long long)F);
    atomicAdd(&g[batch[i] * (long long)F + f], h[idx]);
  }
}

// per-column sum / sum-of-squares -> stats[c], stats[F+c]
__global__ __launch_bounds__(256) void k_colstats(const float* __restrict__ X,
                                                  float* __restrict__ stats, int M, int F) {
  __shared__ float ssum[256];
  __shared__ float ssq[256];
  const int col = blockIdx.x;
  const int tid = threadIdx.x;
  float s = 0.f, q = 0.f;
  for (long long r = (long long)blockIdx.y * blockDim.x + tid; r < M;
       r += (long long)gridDim.y * blockDim.x) {
    float v = X[r * (long long)F + col];
    s += v; q += v * v;
  }
  ssum[tid] = s; ssq[tid] = q;
  __syncthreads();
  for (int off = 128; off > 0; off >>= 1) {
    if (tid < off) { ssum[tid] += ssum[tid + off]; ssq[tid] += ssq[tid + off]; }
    __syncthreads();
  }
  if (tid == 0) {
    atomicAdd(&stats[col],     ssum[0]);
    atomicAdd(&stats[F + col], ssq[0]);
  }
}

// in-place BN (training stats) with optional ReLU
__global__ void k_bn_apply(float* __restrict__ X, const float* __restrict__ stats,
                           const float* __restrict__ gamma, const float* __restrict__ beta,
                           long long Mr, int F, int relu) {
  long long total  = Mr * (long long)F;
  long long stride = (long long)gridDim.x * blockDim.x;
  float invM = 1.0f / (float)Mr;
  for (long long idx = (long long)blockIdx.x * blockDim.x + threadIdx.x; idx < total; idx += stride) {
    int c = (int)(idx % F);
    float mu  = stats[c] * invM;
    float var = stats[F + c] * invM - mu * mu;
    float v = gamma[c] * (X[idx] - mu) * rsqrtf(var + BN_EPS) + beta[c];
    if (relu) v = fmaxf(v, 0.0f);
    X[idx] = v;
  }
}

__global__ void k_concat(const float* __restrict__ a, const float* __restrict__ b,
                         float* __restrict__ c, int M, int Fa, int Fb) {
  int Fc = Fa + Fb;
  long long total  = (long long)M * Fc;
  long long stride = (long long)gridDim.x * blockDim.x;
  for (long long idx = (long long)blockIdx.x * blockDim.x + threadIdx.x; idx < total; idx += stride) {
    long long r = idx / Fc;
    int f = (int)(idx - r * (long long)Fc);
    c[idx] = (f < Fa) ? a[r * Fa + f] : b[r * Fb + (f - Fa)];
  }
}

// out[r] = X[r,:] . w + b   (N=1 head; WMMA pointless here)
__global__ void k_out_head(const float* __restrict__ X, const float* __restrict__ w,
                           const float* __restrict__ b, float* __restrict__ out, int M, int K) {
  int r = blockIdx.x * blockDim.x + threadIdx.x;
  if (r >= M) return;
  float acc = 0.f;
  for (int k = 0; k < K; ++k) acc += X[(long long)r * K + k] * w[k];
  out[r] = acc + b[0];
}

// ---------------------------------------------------------------- bf16 packers
// A[r, 0..Kp): fp32 (+ optional elementwise add) -> bf16, zero pad k >= K
__global__ void k_cvt_a(const float* __restrict__ src, const float* __restrict__ add,
                        __bf16* __restrict__ dst, long long M, int K, int Kp) {
  long long total  = M * (long long)Kp;
  long long stride = (long long)gridDim.x * blockDim.x;
  const bool fuse = (add != nullptr);
  for (long long idx = (long long)blockIdx.x * blockDim.x + threadIdx.x; idx < total; idx += stride) {
    long long r = idx / Kp;
    int c = (int)(idx - r * (long long)Kp);
    float v = 0.f;
    if (c < K) {
      long long i = r * (long long)K + c;
      v = src[i];
      if (fuse) v += add[i];
    }
    dst[idx] = (__bf16)v;
  }
}

// Wt[n*Kp + k] = W[k*N + n] as bf16, zero pad k >= K
__global__ void k_cvt_wt(const float* __restrict__ W, __bf16* __restrict__ Wt,
                         int K, int N, int Kp) {
  long long total  = (long long)N * Kp;
  long long stride = (long long)gridDim.x * blockDim.x;
  for (long long idx = (long long)blockIdx.x * blockDim.x + threadIdx.x; idx < total; idx += stride) {
    int n = (int)(idx / Kp);
    int k = (int)(idx - (long long)n * Kp);
    Wt[idx] = (__bf16)((k < K) ? W[(long long)k * N + n] : 0.f);
  }
}

// ---------------------------------------------------------------- WMMA GEMM
// C = act(A @ W + bias). A: [M, Ka] bf16 (Ka % 64 == 0, zero padded),
// Wt: [N, Ka] bf16 (W transposed, padded). Requires M % 64 == 0, N % 32 == 0
// (true for every GEMM here) -> completely branch/guard-free inner pipeline.
// Block tile 64x32, BK=64 (2 chained WMMAs / step), double-buffered LDS filled
// with CDNA5 async global->LDS copies (ASYNCcnt + split barriers).
__device__ __forceinline__ void async_cp16(unsigned lds_off, const __bf16* gp) {
  asm volatile("global_load_async_to_lds_b128 %0, %1, off"
               :: "v"(lds_off), "v"(gp) : "memory");
}
__device__ __forceinline__ v16bf frag_ld(const __bf16* p) {
  v8bf lo = *(const v8bf*)p;
  v8bf hi = *(const v8bf*)(p + 16);
  return __builtin_shufflevector(lo, hi, 0,1,2,3,4,5,6,7,8,9,10,11,12,13,14,15);
}

__global__ __launch_bounds__(256) void k_gemm_async(const __bf16* __restrict__ A,
                                                    const __bf16* __restrict__ Wt,
                                                    const float* __restrict__ bias,
                                                    float* __restrict__ C,
                                                    __bf16* __restrict__ Cb,
                                                    int M, int N, int Ka,
                                                    int relu, int bf16_out, int ldcb) {
  __shared__ __bf16 As[2][64][64];   // 2 x 8 KB
  __shared__ __bf16 Bs[2][32][64];   // 2 x 4 KB

  const int tid  = threadIdx.x;
  const int lane = tid & 31;
  const int wave = tid >> 5;
  const int wm   = wave & 3;    // M sub-tile
  const int wn   = wave >> 2;   // N sub-tile
  const int half = lane >> 4;   // k-group select (ISA 7.12.2 16-bit layout)
  const int mrow = lane & 15;

  const long long rowBase = (long long)blockIdx.y * 64;
  const int       colBase = blockIdx.x * 32;

  const unsigned asB = (unsigned)(uintptr_t)&As[0][0][0];
  const unsigned bsB = (unsigned)(uintptr_t)&Bs[0][0][0];

  // 16B async chunks: A tile = 512 chunks (2/thread), B tile = 256 (1/thread)
  const int ar0 = tid >> 3;            // chunk tid
  const int ar1 = (tid + 256) >> 3;    // chunk tid+256
  const int acc_ = (tid & 7) * 8;      // bf16 column offset (same for both)
  const int bn_  = tid >> 3;

  const int nsteps = Ka >> 6;
  v8f acc = {};

  // prologue: fill buffer 0 for k0 = 0
  async_cp16(asB + (unsigned)tid * 16u,         A + (rowBase + ar0) * Ka + acc_);
  async_cp16(asB + (unsigned)(tid + 256) * 16u, A + (rowBase + ar1) * Ka + acc_);
  async_cp16(bsB + (unsigned)tid * 16u,         Wt + (size_t)(colBase + bn_) * Ka + acc_);

  for (int s = 0; s < nsteps; ++s) {
    asm volatile("s_wait_asynccnt 0x0" ::: "memory");
    __syncthreads();                       // tile (s&1) visible to all waves
    if (s + 1 < nsteps) {                  // prefetch next tile into other buffer
      const unsigned b  = (unsigned)((s + 1) & 1);
      const int      k0 = (s + 1) << 6;
      async_cp16(asB + b * 8192u + (unsigned)tid * 16u,         A + (rowBase + ar0) * Ka + k0 + acc_);
      async_cp16(asB + b * 8192u + (unsigned)(tid + 256) * 16u, A + (rowBase + ar1) * Ka + k0 + acc_);
      async_cp16(bsB + b * 4096u + (unsigned)tid * 16u,         Wt + (size_t)(colBase + bn_) * Ka + k0 + acc_);
    }
    const int cb = s & 1;
    const __bf16* arow = &As[cb][wm * 16 + mrow][0];
    const __bf16* brow = &Bs[cb][wn * 16 + mrow][0];
#pragma unroll
    for (int t = 0; t < 2; ++t) {
      const int kb = t * 32 + half * 8;
      v16bf af  = frag_ld(arow + kb);
      v16bf bfv = frag_ld(brow + kb);
      acc = __builtin_amdgcn_wmma_f32_16x16x32_bf16(false, af, false, bfv,
                                                    (short)0, acc, false, false);
    }
    __syncthreads();                       // all reads of tile (s&1) done
  }

  // D layout: VGPR j of lane L -> M = j + 8*(L>>4), N = L&15
  const int gc = colBase + wn * 16 + mrow;
  const float bv = bias[gc];
#pragma unroll
  for (int j = 0; j < 8; ++j) {
    long long gr = rowBase + wm * 16 + half * 8 + j;
    float v = acc[j] + bv;
    if (relu) v = fmaxf(v, 0.0f);
    if (bf16_out) {
      Cb[gr * ldcb + gc] = (__bf16)v;
      for (int c2 = gc + N; c2 < ldcb; c2 += N)   // zero K-padding for next GEMM
        Cb[gr * ldcb + c2] = (__bf16)0.f;
    } else {
      C[gr * N + gc] = v;
    }
  }
}

// ---------------------------------------------------------------- launcher
extern "C" void kernel_launch(void* const* d_in, const int* in_sizes, int n_in,
                              void* d_out, int out_size, void* d_ws, size_t ws_size,
                              hipStream_t stream) {
  const float*     x      = (const float*)d_in[0];
  const long long* eidx   = (const long long*)d_in[1];   // [2,E] int64
  const long long* batch  = (const long long*)d_in[2];   // [N]  int64
  const float*     target = (const float*)d_in[3];       // [B,4096]

  const long long Nn  = in_sizes[2];
  const long long E   = in_sizes[1] / 2;
  const int T_IN = 4096;
  const int B    = in_sizes[3] / T_IN;
  const int F_IN = (int)(in_sizes[0] / Nn);
  const int DIM  = 32;

  int p = 4;
  const float *gw1[5], *gb1[5], *gw2[5], *gb2[5], *gga[5], *gbe[5];
  for (int i = 0; i < 5; ++i) {
    gw1[i] = (const float*)d_in[p++]; gb1[i] = (const float*)d_in[p++];
    gw2[i] = (const float*)d_in[p++]; gb2[i] = (const float*)d_in[p++];
    gga[i] = (const float*)d_in[p++]; gbe[i] = (const float*)d_in[p++];
  }
  const float* fc1xd_w = (const float*)d_in[p++];
  const float* fc1xd_b = (const float*)d_in[p++];
  const float *xtw[3], *xtb[3], *xtg[3], *xtbt[3];
  for (int i = 0; i < 3; ++i) {
    xtw[i]  = (const float*)d_in[p++]; xtb[i]  = (const float*)d_in[p++];
    xtg[i]  = (const float*)d_in[p++]; xtbt[i] = (const float*)d_in[p++];
  }
  const float* fc1_w = (const float*)d_in[p++]; const float* fc1_b = (const float*)d_in[p++];
  const float* fc2_w = (const float*)d_in[p++]; const float* fc2_b = (const float*)d_in[p++];
  const float* outw  = (const float*)d_in[p++]; const float* outb  = (const float*)d_in[p++];

  // ---- bump allocator over d_ws ----
  char* ws = (char*)d_ws;
  size_t off = 0;
  auto walloc = [&](size_t bytes) -> void* {
    void* r = (void*)(ws + off);
    off += (bytes + 255) & ~(size_t)255;
    return r;
  };
  size_t aggBytes = ((size_t)Nn * F_IN * 4 + 255) & ~(size_t)255;
  float*  agg   = (float*)walloc((size_t)Nn * F_IN * 4);
  float*  hA    = (float*)walloc((size_t)Nn * DIM * 4);
  float*  hB    = (float*)walloc((size_t)Nn * DIM * 4);
  float*  stats = (float*)walloc(2 * 512 * 4);
  float*  g     = (float*)walloc((size_t)B * DIM * 4);
  float*  gfc   = (float*)walloc((size_t)B * 128 * 4);
  // bf16 staging: Abf must hold max(Nn x 128, B x 4096) elements
  size_t abfElems = (size_t)Nn * 128;
  size_t tElems   = (size_t)B * T_IN;
  if (tElems > abfElems) abfElems = tElems;
  __bf16* Abf = (__bf16*)walloc(abfElems * 2);
  __bf16* Mbf = (__bf16*)walloc((size_t)Nn * 64 * 2);     // GIN hidden, Kp=64
  __bf16* Wt  = (__bf16*)walloc((size_t)1024 * 4096 * 2 / 2); // 512*4096 max -> 4.2MB

  // Overlay protein/head fp32 buffers on the agg region (dead after graph stage)
  size_t need = 0;
  auto osz = [&](size_t b) { size_t r = need; need += (b + 255) & ~(size_t)255; return r; };
  size_t o_xt1 = osz((size_t)B * 512 * 4), o_xt2 = osz((size_t)B * 256 * 4),
         o_xt3 = osz((size_t)B * 128 * 4), o_cat = osz((size_t)B * 256 * 4),
         o_fcA = osz((size_t)B * 1024 * 4), o_fcB = osz((size_t)B * 256 * 4);
  char* ovl = (need <= aggBytes) ? (char*)agg : (char*)walloc(need);
  float* xt1 = (float*)(ovl + o_xt1);
  float* xt2 = (float*)(ovl + o_xt2);
  float* xt3 = (float*)(ovl + o_xt3);
  float* cat = (float*)(ovl + o_cat);
  float* fcA = (float*)(ovl + o_fcA);
  float* fcB = (float*)(ovl + o_fcB);

  const dim3 blk(256);
  auto cvtA = [&](const float* src, const float* add, __bf16* dst, long long M, int K, int Kp) {
    long long tot = M * (long long)Kp;
    unsigned nb = (unsigned)((tot + 255) / 256); if (nb > 8192u) nb = 8192u;
    k_cvt_a<<<dim3(nb), blk, 0, stream>>>(src, add, dst, M, K, Kp);
  };
  auto cvtW = [&](const float* W, int K, int N, int Kp) {
    long long tot = (long long)N * Kp;
    unsigned nb = (unsigned)((tot + 255) / 256); if (nb > 4096u) nb = 4096u;
    k_cvt_wt<<<dim3(nb), blk, 0, stream>>>(W, Wt, K, N, Kp);
  };
  auto gemm = [&](const __bf16* A, const float* bias, float* C, __bf16* Cb,
                  int M, int N, int Ka, int relu, int bf16_out, int ldcb) {
    dim3 grd((unsigned)(N / 32), (unsigned)(M / 64));
    k_gemm_async<<<grd, blk, 0, stream>>>(A, Wt, bias, C, Cb, M, N, Ka, relu, bf16_out, ldcb);
  };
  auto bnorm = [&](float* X, int M, int F, const float* ga, const float* be, int relu) {
    k_zero<<<dim3(1), blk, 0, stream>>>(stats, 2LL * F);
    int chunks = (M > 65536) ? 40 : 16;
    k_colstats<<<dim3(F, chunks), blk, 0, stream>>>(X, stats, M, F);
    long long tot = (long long)M * F;
    unsigned nb = (unsigned)((tot + 255) / 256); if (nb > 8192u) nb = 8192u;
    k_bn_apply<<<dim3(nb), blk, 0, stream>>>(X, stats, ga, be, M, F, relu);
  };
  auto kpad = [](int K) { return (K + 63) & ~63; };

  // ---- GIN stack ----
  const float* hin = x;
  int Fin = F_IN;
  float* hout = hA;
  for (int i = 0; i < 5; ++i) {
    int Kp = kpad(Fin);
    k_zero<<<dim3(8192), blk, 0, stream>>>(agg, Nn * (long long)Fin);
    k_scatter_add<<<dim3(16384), blk, 0, stream>>>(hin, eidx, eidx + E, agg, E, Fin);
    cvtA(hin, agg, Abf, Nn, Fin, Kp);                      // A = h + sum_neigh(h)
    cvtW(gw1[i], Fin, DIM, Kp);
    gemm(Abf, gb1[i], nullptr, Mbf, (int)Nn, DIM, Kp, 1, 1, 64); // relu -> bf16 hidden
    cvtW(gw2[i], DIM, DIM, 64);
    gemm(Mbf, gb2[i], hout, nullptr, (int)Nn, DIM, 64, 1, 0, 0); // relu -> fp32
    bnorm(hout, (int)Nn, DIM, gga[i], gbe[i], 0);
    hin = hout; Fin = DIM;
    hout = (hout == hA) ? hB : hA;
  }

  // ---- pool + fc1_xd ----
  k_zero<<<dim3(512), blk, 0, stream>>>(g, (long long)B * DIM);
  k_pool<<<dim3(8192), blk, 0, stream>>>(hin, batch, g, Nn, DIM);
  cvtA(g, nullptr, Abf, B, DIM, 64);
  cvtW(fc1xd_w, DIM, 128, 64);
  gemm(Abf, fc1xd_b, gfc, nullptr, B, 128, 64, 1, 0, 0);

  // ---- protein tower: linear -> BN -> relu ----
  cvtA(target, nullptr, Abf, B, T_IN, T_IN);
  cvtW(xtw[0], T_IN, 512, T_IN);
  gemm(Abf, xtb[0], xt1, nullptr, B, 512, T_IN, 0, 0, 0);
  bnorm(xt1, B, 512, xtg[0], xtbt[0], 1);
  cvtA(xt1, nullptr, Abf, B, 512, 512);
  cvtW(xtw[1], 512, 256, 512);
  gemm(Abf, xtb[1], xt2, nullptr, B, 256, 512, 0, 0, 0);
  bnorm(xt2, B, 256, xtg[1], xtbt[1], 1);
  cvtA(xt2, nullptr, Abf, B, 256, 256);
  cvtW(xtw[2], 256, 128, 256);
  gemm(Abf, xtb[2], xt3, nullptr, B, 128, 256, 0, 0, 0);
  bnorm(xt3, B, 128, xtg[2], xtbt[2], 1);

  // ---- head ----
  k_concat<<<dim3(2048), blk, 0, stream>>>(gfc, xt3, cat, B, 128, 128);
  cvtA(cat, nullptr, Abf, B, 256, 256);
  cvtW(fc1_w, 256, 1024, 256);
  gemm(Abf, fc1_b, fcA, nullptr, B, 1024, 256, 1, 0, 0);
  cvtA(fcA, nullptr, Abf, B, 1024, 1024);
  cvtW(fc2_w, 1024, 256, 1024);
  gemm(Abf, fc2_b, fcB, nullptr, B, 256, 1024, 1, 0, 0);
  k_out_head<<<dim3((B + 255) / 256), blk, 0, stream>>>(fcB, outw, outb, (float*)d_out, B, 256);

  (void)n_in; (void)out_size; (void)ws_size;
}
